// SimpleHeteroGNN_79517024519044
// MI455X (gfx1250) — compile-verified, matrix operand order checked by reading, not run on hardware
//
#include <hip/hip_runtime.h>
#include <hip/hip_bf16.h>

typedef __attribute__((ext_vector_type(16))) _Float16 v16h;
typedef __attribute__((ext_vector_type(8)))  float    v8f;

#define H_DIM 128

// ---------------------------------------------------------------------------
// zero fill
// ---------------------------------------------------------------------------
__global__ void zero_f32(float* __restrict__ p, size_t n) {
    size_t i = (size_t)blockIdx.x * blockDim.x + threadIdx.x;
    size_t stride = (size_t)gridDim.x * blockDim.x;
    for (; i < n; i += stride) p[i] = 0.0f;
}

// ---------------------------------------------------------------------------
// segment-sum scatter: agg[dst[e], :] += xsrc[src[e], :],  cnt[dst[e]] += 1
// (f32 master activations; f32 atomics for exact segment mean)
// ---------------------------------------------------------------------------
__global__ void scatter_add_mean(const float* __restrict__ xsrc,
                                 const int* __restrict__ src,
                                 const int* __restrict__ dst,
                                 float* __restrict__ agg,
                                 float* __restrict__ cnt,
                                 int nedges) {
    size_t t = (size_t)blockIdx.x * blockDim.x + threadIdx.x;
    size_t total = (size_t)nedges * 32;
    size_t stride = (size_t)gridDim.x * blockDim.x;
    for (; t < total; t += stride) {
        int e = (int)(t >> 5);
        int c = (int)(t & 31);
        int s = src[e], d = dst[e];
        float4 v = *(const float4*)(xsrc + (size_t)s * H_DIM + c * 4);
        float* ap = agg + (size_t)d * H_DIM + c * 4;
        atomicAdd(ap + 0, v.x);
        atomicAdd(ap + 1, v.y);
        atomicAdd(ap + 2, v.z);
        atomicAdd(ap + 3, v.w);
        if (c == 0) atomicAdd(cnt + d, 1.0f);
    }
}

// ---------------------------------------------------------------------------
// Pack [W1;W2] ([Ktot,128] f32 row-major) into WMMA B-fragment order, f16:
//   out[ ((nt*T + t)*32 + lane)*16 + j ]  with
//     K = t*32 + (lane>>4)*8 + (j<8 ? j : 8+j),  N = nt*16 + (lane&15)
// ---------------------------------------------------------------------------
__global__ void pack_b_frags(const float* __restrict__ W1, int K1,
                             const float* __restrict__ W2, int K2,
                             _Float16* __restrict__ out) {
    const int Ktot = K1 + K2;
    const int T = Ktot >> 5;
    int total = Ktot * H_DIM;
    for (int o = blockIdx.x * blockDim.x + threadIdx.x; o < total;
         o += gridDim.x * blockDim.x) {
        int j  = o & 15;
        int l  = (o >> 4) & 31;
        int t  = (o >> 9) % T;
        int nt = o / (T << 9);
        int k  = t * 32 + ((l >> 4) << 3) + (j < 8 ? j : 8 + j);
        int n  = nt * 16 + (l & 15);
        float v = (k < K1) ? W1[k * H_DIM + n] : W2[(k - K1) * H_DIM + n];
        out[o] = (_Float16)v;
    }
}

// ---------------------------------------------------------------------------
// out = relu( [A1*rowscale | A2] @ Bpack + bias )
//   A1: f32 (agg / raw features; optional per-row 1/max(cnt,1) scale)
//   A2: f16 activation shadow (direct fragment loads, no converts)
// Writes both f32 master copy (scatter source) and f16 shadow (next layer).
// Two independent 16-row tiles per wave iteration; B register-resident.
// nrows % 32 == 0.
// ---------------------------------------------------------------------------
template <int K1, int K2, bool SCALE>
__global__ __launch_bounds__(256)
void wmma_gemm_v4(const float* __restrict__ A1, const float* __restrict__ cnt,
                  const _Float16* __restrict__ A2,
                  const _Float16* __restrict__ Bpack,
                  const float* __restrict__ bias,
                  float* __restrict__ out32, _Float16* __restrict__ out16,
                  int nrows) {
    constexpr int Ktot = K1 + K2;
    constexpr int T = Ktot / 32;

    const int lane = threadIdx.x & 31;
    const int wave = threadIdx.x >> 5;
    const int hf   = lane >> 4;
    const int mr   = lane & 15;
    const int ncol = wave * 16 + mr;

    v16h bfrag[T];
#pragma unroll
    for (int t = 0; t < T; ++t)
        bfrag[t] = *(const v16h*)(Bpack + (((size_t)wave * T + t) * 32 + lane) * 16);

    const float bv = bias[ncol];
    const int npairs = nrows >> 5;
    for (int pair = blockIdx.x; pair < npairs; pair += gridDim.x) {
        const int row0 = pair * 32 + mr;
        const int row1 = row0 + 16;
        const float s0 = SCALE ? (1.0f / fmaxf(cnt[row0], 1.0f)) : 1.0f;
        const float s1 = SCALE ? (1.0f / fmaxf(cnt[row1], 1.0f)) : 1.0f;
        const float* r0A1 = A1 + (size_t)row0 * K1;
        const float* r1A1 = A1 + (size_t)row1 * K1;
        const _Float16* r0A2 = (K2 > 0) ? (A2 + (size_t)row0 * K2) : nullptr;
        const _Float16* r1A2 = (K2 > 0) ? (A2 + (size_t)row1 * K2) : nullptr;

        v8f c0 = {}, c1 = {};
#pragma unroll
        for (int t = 0; t < T; ++t) {
            const int k0 = t * 32;
            v16h a0, a1;
            if (K1 == Ktot || k0 < K1) {
                const float* p0 = r0A1 + k0 + hf * 8;
                const float* p1 = r1A1 + k0 + hf * 8;
#pragma unroll
                for (int j = 0; j < 8; ++j) {
                    a0[j]     = (_Float16)(p0[j] * s0);
                    a0[j + 8] = (_Float16)(p0[16 + j] * s0);
                    a1[j]     = (_Float16)(p1[j] * s1);
                    a1[j + 8] = (_Float16)(p1[16 + j] * s1);
                }
            } else {
                const _Float16* p0 = r0A2 + (k0 - K1) + hf * 8;
                const _Float16* p1 = r1A2 + (k0 - K1) + hf * 8;
#pragma unroll
                for (int j = 0; j < 8; ++j) {
                    a0[j]     = p0[j];
                    a0[j + 8] = p0[16 + j];
                    a1[j]     = p1[j];
                    a1[j + 8] = p1[16 + j];
                }
            }
            c0 = __builtin_amdgcn_wmma_f32_16x16x32_f16(
                    false, a0, false, bfrag[t], (short)0, c0, false, false);
            c1 = __builtin_amdgcn_wmma_f32_16x16x32_f16(
                    false, a1, false, bfrag[t], (short)0, c1, false, false);
        }
        const int mb0 = pair * 32 + hf * 8;
        const int mb1 = mb0 + 16;
#pragma unroll
        for (int r = 0; r < 8; ++r) {
            float v0 = fmaxf(c0[r] + bv, 0.0f);
            float v1 = fmaxf(c1[r] + bv, 0.0f);
            out32[(size_t)(mb0 + r) * H_DIM + ncol] = v0;
            out32[(size_t)(mb1 + r) * H_DIM + ncol] = v1;
            out16[(size_t)(mb0 + r) * H_DIM + ncol] = (_Float16)v0;
            out16[(size_t)(mb1 + r) * H_DIM + ncol] = (_Float16)v1;
        }
    }
}

// ---------------------------------------------------------------------------
// Edge predictor on f16 node tables (L2-resident: 25.6MB + 5.1MB):
//   h = relu(BN([ux[s]||mx[d]] @ p1_W + p1_b))
//   out = sigmoid(h.p2_W + p2_b)*4 + 1
// Two 16-edge tiles per iteration; zero converts in the hot loop;
// shfl_xor butterfly reduction + 32-float LDS accumulator.
// ---------------------------------------------------------------------------
__global__ __launch_bounds__(256)
void edge_mlp_v4(const _Float16* __restrict__ ux, const _Float16* __restrict__ mx,
                 const int* __restrict__ esrc, const int* __restrict__ edst,
                 const _Float16* __restrict__ p1pack, const float* __restrict__ p1b,
                 const float* __restrict__ gamma, const float* __restrict__ beta,
                 const float* __restrict__ mean,  const float* __restrict__ var,
                 const float* __restrict__ p2W,   const float* __restrict__ p2b,
                 float* __restrict__ out, int nedges) {
    __shared__ float acc[32];
    constexpr int T = 8;   // K = 256

    const int lane = threadIdx.x & 31;
    const int wave = threadIdx.x >> 5;
    const int hf   = lane >> 4;
    const int mr   = lane & 15;
    const int ncol = wave * 16 + mr;

    v16h bfrag[T];
#pragma unroll
    for (int t = 0; t < T; ++t)
        bfrag[t] = *(const v16h*)(p1pack + (((size_t)wave * T + t) * 32 + lane) * 16);

    const float g   = gamma[ncol];
    const float bt  = beta[ncol];
    const float mu  = mean[ncol];
    const float inv = rsqrtf(var[ncol] + 1e-5f);
    const float b1  = p1b[ncol];
    const float w2  = p2W[ncol];
    const float b2  = p2b[0];

    const int npairs = nedges >> 5;
    for (int pair = blockIdx.x; pair < npairs; pair += gridDim.x) {
        if (threadIdx.x < 32) acc[threadIdx.x] = 0.0f;
        __syncthreads();

        const int e0 = pair * 32 + mr;
        const int e1 = e0 + 16;
        const _Float16* r0u = ux + (size_t)esrc[e0] * H_DIM;
        const _Float16* r0m = mx + (size_t)edst[e0] * H_DIM;
        const _Float16* r1u = ux + (size_t)esrc[e1] * H_DIM;
        const _Float16* r1m = mx + (size_t)edst[e1] * H_DIM;

        v8f c0 = {}, c1 = {};
#pragma unroll
        for (int t = 0; t < T; ++t) {
            const _Float16* a0p = ((t < 4) ? (r0u + t * 32) : (r0m + (t - 4) * 32)) + hf * 8;
            const _Float16* a1p = ((t < 4) ? (r1u + t * 32) : (r1m + (t - 4) * 32)) + hf * 8;
            v16h a0, a1;
#pragma unroll
            for (int j = 0; j < 8; ++j) {
                a0[j]     = a0p[j];
                a0[j + 8] = a0p[16 + j];
                a1[j]     = a1p[j];
                a1[j + 8] = a1p[16 + j];
            }
            c0 = __builtin_amdgcn_wmma_f32_16x16x32_f16(
                    false, a0, false, bfrag[t], (short)0, c0, false, false);
            c1 = __builtin_amdgcn_wmma_f32_16x16x32_f16(
                    false, a1, false, bfrag[t], (short)0, c1, false, false);
        }
        // BN + relu + column weight, reduce over the wave's 16 columns
        float v0[8], v1[8];
#pragma unroll
        for (int r = 0; r < 8; ++r) {
            float h0 = (c0[r] + b1 - mu) * inv * g + bt;
            float h1 = (c1[r] + b1 - mu) * inv * g + bt;
            v0[r] = fmaxf(h0, 0.0f) * w2;
            v1[r] = fmaxf(h1, 0.0f) * w2;
        }
#pragma unroll
        for (int off = 1; off < 16; off <<= 1)
#pragma unroll
            for (int r = 0; r < 8; ++r) {
                v0[r] += __shfl_xor(v0[r], off, 32);
                v1[r] += __shfl_xor(v1[r], off, 32);
            }
        if (mr == 0) {
#pragma unroll
            for (int r = 0; r < 8; ++r) {
                atomicAdd(&acc[hf * 8 + r],      v0[r]);
                atomicAdd(&acc[16 + hf * 8 + r], v1[r]);
            }
        }
        __syncthreads();
        if (threadIdx.x < 32) {
            float h2 = acc[threadIdx.x] + b2;
            float sg = 1.0f / (1.0f + __expf(-h2));
            out[pair * 32 + threadIdx.x] = sg * 4.0f + 1.0f;
        }
        __syncthreads();
    }
}

// ---------------------------------------------------------------------------
extern "C" void kernel_launch(void* const* d_in, const int* in_sizes, int n_in,
                              void* d_out, int out_size, void* d_ws, size_t ws_size,
                              hipStream_t stream) {
    (void)in_sizes; (void)n_in; (void)out_size; (void)ws_size;
    const int NU = 100000, NM = 20000, E = 1000000, EUU = 500000, H = H_DIM;

    const float* user_feat  = (const float*)d_in[0];
    const float* movie_feat = (const float*)d_in[1];
    const float* W_user  = (const float*)d_in[2];
    const float* b_user  = (const float*)d_in[3];
    const float* W_movie = (const float*)d_in[4];
    const float* b_movie = (const float*)d_in[5];
    const float* Wl[5] = { (const float*)d_in[6],  (const float*)d_in[9],
                           (const float*)d_in[12], (const float*)d_in[15],
                           (const float*)d_in[18] };
    const float* bl[5] = { (const float*)d_in[7],  (const float*)d_in[10],
                           (const float*)d_in[13], (const float*)d_in[16],
                           (const float*)d_in[19] };
    const float* Wr[5] = { (const float*)d_in[8],  (const float*)d_in[11],
                           (const float*)d_in[14], (const float*)d_in[17],
                           (const float*)d_in[20] };
    const float* p1_W = (const float*)d_in[21];
    const float* p1_b = (const float*)d_in[22];
    const float* bn_g = (const float*)d_in[23];
    const float* bn_b = (const float*)d_in[24];
    const float* bn_m = (const float*)d_in[25];
    const float* bn_v = (const float*)d_in[26];
    const float* p2_W = (const float*)d_in[27];
    const float* p2_b = (const float*)d_in[28];
    const int* rated_src = (const int*)d_in[29];
    const int* rated_dst = (const int*)d_in[30];
    const int* uu_src    = (const int*)d_in[31];
    const int* uu_dst    = (const int*)d_in[32];
    float* outp = (float*)d_out;

    // workspace carve-up
    char* ws = (char*)d_ws;
    float* ux_a = (float*)ws; ws += (size_t)NU * H * sizeof(float);
    float* ux_b = (float*)ws; ws += (size_t)NU * H * sizeof(float);
    float* mx_a = (float*)ws; ws += (size_t)NM * H * sizeof(float);
    float* mx_b = (float*)ws; ws += (size_t)NM * H * sizeof(float);
    float* agg  = (float*)ws; ws += (size_t)NU * H * sizeof(float);
    float* cnt  = (float*)ws; ws += (size_t)NU * sizeof(float);
    _Float16* uxh_a = (_Float16*)ws; ws += (size_t)NU * H * sizeof(_Float16);
    _Float16* uxh_b = (_Float16*)ws; ws += (size_t)NU * H * sizeof(_Float16);
    _Float16* mxh_a = (_Float16*)ws; ws += (size_t)NM * H * sizeof(_Float16);
    _Float16* mxh_b = (_Float16*)ws; ws += (size_t)NM * H * sizeof(_Float16);
    _Float16* pk_user  = (_Float16*)ws; ws += (size_t)64  * H * sizeof(_Float16);
    _Float16* pk_movie = (_Float16*)ws; ws += (size_t)128 * H * sizeof(_Float16);
    _Float16* pk_sage[5];
    for (int i = 0; i < 5; ++i) { pk_sage[i] = (_Float16*)ws; ws += (size_t)256 * H * sizeof(_Float16); }
    _Float16* pk_p1 = (_Float16*)ws; ws += (size_t)256 * H * sizeof(_Float16);

    const dim3 blk(256);
    const int gU = 3125;   // NU/32 pairs exactly
    const int gM = 625;    // NM/32 pairs exactly
    const int gE = 4096;   // E/32 = 31250 pairs, grid-stride
    const int gScat = 8192, gZero = 2048, gPack = 64;

    // ---- pack all weights into WMMA B-fragment f16 order ----
    pack_b_frags<<<gPack, blk, 0, stream>>>(W_user, 64, nullptr, 0, pk_user);
    pack_b_frags<<<gPack, blk, 0, stream>>>(W_movie, 128, nullptr, 0, pk_movie);
    for (int i = 0; i < 5; ++i)
        pack_b_frags<<<gPack, blk, 0, stream>>>(Wl[i], 128, Wr[i], 128, pk_sage[i]);
    pack_b_frags<<<gPack, blk, 0, stream>>>(p1_W, 256, nullptr, 0, pk_p1);

    // ---- node encoders ----
    wmma_gemm_v4<64, 0, false><<<gU, blk, 0, stream>>>(
        user_feat, nullptr, nullptr, pk_user, b_user, ux_a, uxh_a, NU);
    wmma_gemm_v4<128, 0, false><<<gM, blk, 0, stream>>>(
        movie_feat, nullptr, nullptr, pk_movie, b_movie, mx_a, mxh_a, NM);

    // ---- u2m1: users -> movies ----
    zero_f32<<<gZero, blk, 0, stream>>>(agg, (size_t)NM * H);
    zero_f32<<<gZero, blk, 0, stream>>>(cnt, (size_t)NM);
    scatter_add_mean<<<gScat, blk, 0, stream>>>(ux_a, rated_src, rated_dst, agg, cnt, E);
    wmma_gemm_v4<128, 128, true><<<gM, blk, 0, stream>>>(
        agg, cnt, mxh_a, pk_sage[0], bl[0], mx_b, mxh_b, NM);

    // ---- m2u1: movies -> users ----
    zero_f32<<<gZero, blk, 0, stream>>>(agg, (size_t)NU * H);
    zero_f32<<<gZero, blk, 0, stream>>>(cnt, (size_t)NU);
    scatter_add_mean<<<gScat, blk, 0, stream>>>(mx_b, rated_dst, rated_src, agg, cnt, E);
    wmma_gemm_v4<128, 128, true><<<gU, blk, 0, stream>>>(
        agg, cnt, uxh_a, pk_sage[1], bl[1], ux_b, uxh_b, NU);

    // ---- u2m2 ----
    zero_f32<<<gZero, blk, 0, stream>>>(agg, (size_t)NM * H);
    zero_f32<<<gZero, blk, 0, stream>>>(cnt, (size_t)NM);
    scatter_add_mean<<<gScat, blk, 0, stream>>>(ux_b, rated_src, rated_dst, agg, cnt, E);
    wmma_gemm_v4<128, 128, true><<<gM, blk, 0, stream>>>(
        agg, cnt, mxh_b, pk_sage[2], bl[2], mx_a, mxh_a, NM);

    // ---- m2u2 ----
    zero_f32<<<gZero, blk, 0, stream>>>(agg, (size_t)NU * H);
    zero_f32<<<gZero, blk, 0, stream>>>(cnt, (size_t)NU);
    scatter_add_mean<<<gScat, blk, 0, stream>>>(mx_a, rated_dst, rated_src, agg, cnt, E);
    wmma_gemm_v4<128, 128, true><<<gU, blk, 0, stream>>>(
        agg, cnt, uxh_b, pk_sage[3], bl[3], ux_a, uxh_a, NU);

    // ---- u2u ----
    zero_f32<<<gZero, blk, 0, stream>>>(agg, (size_t)NU * H);
    zero_f32<<<gZero, blk, 0, stream>>>(cnt, (size_t)NU);
    scatter_add_mean<<<gScat, blk, 0, stream>>>(ux_a, uu_src, uu_dst, agg, cnt, EUU);
    wmma_gemm_v4<128, 128, true><<<gU, blk, 0, stream>>>(
        agg, cnt, uxh_a, pk_sage[4], bl[4], ux_b, uxh_b, NU);

    // ---- edge predictor (f16 node tables) -> d_out ----
    edge_mlp_v4<<<gE, blk, 0, stream>>>(
        uxh_b, mxh_a, rated_src, rated_dst, pk_p1, p1_b,
        bn_g, bn_b, bn_m, bn_v, p2_W, p2_b, outp, E);
}
